// LSTM_63488206569533
// MI455X (gfx1250) — matrix operand order repaired and checked by reading, not compiled, not run on hardware
//
#include <hip/hip_runtime.h>

// ---------------- problem constants ----------------
#define Tdim   1024
#define Bdim   128
#define Idim   256
#define Hdim   512
#define Kdim   768            // Hdim + Idim (concat [h | x_t])
#define NBLK   32             // 512 hidden cols / 16 per block
#define THREADS 256           // 8 wave32 per block
#define CHUNKS 96             // 768 bf16 elems / 8 per 16B chunk
#define ROWU4  97             // padded row length in uint4 (776 bf16) -> bank spread

typedef __attribute__((ext_vector_type(16))) __bf16    v16bf;
typedef __attribute__((ext_vector_type(8)))  float     v8f;
typedef __attribute__((ext_vector_type(4)))  unsigned  uv4;
typedef __attribute__((ext_vector_type(4)))  float     fv4;

// exact payload type expected by the async-to-LDS builtin (from compiler diag):
// '__attribute__((__vector_size__(4 * sizeof(int)))) int'
typedef int b128_t __attribute__((vector_size(16)));

#define GLOBAL_AS __attribute__((address_space(1)))
#define LDS_AS    __attribute__((address_space(3)))

union FragBF { uv4 u[2]; v16bf v; };

// ---------------- helpers ----------------
__device__ __forceinline__ unsigned short f2bf(float f) {
    unsigned u = __float_as_uint(f);
    u += 0x7FFFu + ((u >> 16) & 1u);          // round-to-nearest-even
    return (unsigned short)(u >> 16);
}
__device__ __forceinline__ float bf2f(unsigned short s) {
    return __uint_as_float(((unsigned)s) << 16);
}
__device__ __forceinline__ float sigm(float x)  { return 1.0f / (1.0f + __expf(-x)); }
__device__ __forceinline__ float tanh_(float x) { return 1.0f - 2.0f / (__expf(2.0f * x) + 1.0f); }

// ---------------- workspace layout (bytes) ----------------
// [0,256)          : barrier {count, generation}
// [256, +131072)   : h state, bf16 [128][512]
// [131584, +8192)  : fused bias f32 [2048]  (b_ih + b_hh)
// [139776, +3.1MB) : Wcat bf16 [2048][768]  (row n: [W_hh[n][0:512] | W_ih[n][0:256]])
#define OFF_H    256
#define OFF_BIAS 131584
#define OFF_WCAT 139776

// ---------------- init: zero h state + barrier ----------------
__global__ void k_init(unsigned* bar, uv4* h) {
    int i = blockIdx.x * blockDim.x + threadIdx.x;
    uv4 z = {0u, 0u, 0u, 0u};
    if (i < (Bdim * Hdim * 2) / 16) h[i] = z;     // 131072 B = 8192 uv4
    if (i < 2) bar[i] = 0u;
}

// ---------------- prep: repack weights to bf16 Wcat + fuse biases ----------------
__global__ void k_prep_w(const float* __restrict__ Wih, const float* __restrict__ Whh,
                         const float* __restrict__ bih, const float* __restrict__ bhh,
                         unsigned short* __restrict__ wcat, float* __restrict__ bias) {
    const long stride = (long)gridDim.x * blockDim.x;
    const long tid0   = (long)blockIdx.x * blockDim.x + threadIdx.x;
    for (long idx = tid0; idx < (long)4 * Hdim * Kdim; idx += stride) {
        int  k = (int)(idx % Kdim);
        long n = idx / Kdim;
        float v = (k < Hdim) ? Whh[n * Hdim + k] : Wih[n * Idim + (k - Hdim)];
        wcat[idx] = f2bf(v);
    }
    for (long n = tid0; n < 4 * Hdim; n += stride) bias[n] = bih[n] + bhh[n];
}

// ---------------- persistent recurrent LSTM kernel ----------------
// Block bid owns hidden columns [16*bid, 16*bid+16): computes gates i,f,g,o for
// those columns over all 128 batch rows, then the cell update. Steps are
// separated by a device-wide barrier (h must be globally visible).
__global__ void __launch_bounds__(THREADS, 1)
k_lstm(const float* __restrict__ x,              // [B][T][I] f32
       const unsigned short* __restrict__ wcat,  // [2048][768] bf16
       const float* __restrict__ bias,           // [2048] f32
       unsigned short* __restrict__ hstate,      // [128][512] bf16
       unsigned* __restrict__ bar)
{
    extern __shared__ __align__(16) char smem[];
    uv4* wlds = (uv4*)smem;                          // 64 rows x 97 uv4  (96 KB+pad)
    uv4* alds = (uv4*)(smem + 64 * ROWU4 * 16);      // 128 rows x 97 uv4 (194 KB+pad)

    const int tid   = threadIdx.x;
    const int bid   = blockIdx.x;
    const int wave  = tid >> 5;
    const int lane  = tid & 31;
    const int lhalf = lane >> 4;       // K-half select per ISA 16-bit operand layout
    const int l16   = lane & 15;

    // ---- stage this block's 64 weight rows (4 gates x 16 cols) into LDS, once ----
    // Async DMA path: memory -> LDS directly, no VGPR round trip (ASYNCcnt).
    const uv4* wsrc = (const uv4*)wcat;              // 96 uv4 per 768-elem row
    for (int idx = tid; idx < 64 * CHUNKS; idx += THREADS) {
        int rowL = idx / CHUNKS, c = idx % CHUNKS;
        int g = rowL >> 4, j = rowL & 15;
        long nglob = (long)g * Hdim + bid * 16 + j;
        __builtin_amdgcn_global_load_async_to_lds_b128(
            (GLOBAL_AS b128_t*)(wsrc + nglob * CHUNKS + c),
            (LDS_AS    b128_t*)(wlds + rowL * ROWU4 + c), 0, 0);
    }

    // per-lane bias for this lane's output column, per gate
    float brs[4];
#pragma unroll
    for (int g = 0; g < 4; ++g) brs[g] = bias[g * Hdim + bid * 16 + l16];

    // cell state lives in registers: lane owns (row = wave*16 + r + 8*lhalf, col)
    float creg[8];
#pragma unroll
    for (int r = 0; r < 8; ++r) creg[r] = 0.0f;

    const uv4* hsrc4 = (const uv4*)hstate;           // 64 uv4 per 512-elem row
    __builtin_amdgcn_s_wait_asynccnt(0);
    __syncthreads();

#pragma unroll 1
    for (int t = 0; t < Tdim; ++t) {
        // ---- stage A = [h | x_t] (128 x 768 bf16) into LDS ----
        // h tile: async global->LDS DMA (bf16 already), overlaps with x conversion.
        for (int idx = tid; idx < Bdim * 64; idx += THREADS) {
            int row = idx >> 6, c = idx & 63;
            __builtin_amdgcn_global_load_async_to_lds_b128(
                (GLOBAL_AS b128_t*)(hsrc4 + row * 64 + c),
                (LDS_AS    b128_t*)(alds + row * ROWU4 + c), 0, 0);
        }
        // x_t slice: f32 -> bf16 conversion through VGPRs (runs while DMA in flight)
        for (int idx = tid; idx < Bdim * 32; idx += THREADS) {
            int row = idx >> 5, c = idx & 31;
            const float* xr = x + ((long)row * Tdim + t) * Idim + c * 8;
            fv4 x0 = ((const fv4*)xr)[0];
            fv4 x1 = ((const fv4*)xr)[1];
            union { unsigned short s[8]; uv4 u; } p;
            p.s[0] = f2bf(x0[0]); p.s[1] = f2bf(x0[1]);
            p.s[2] = f2bf(x0[2]); p.s[3] = f2bf(x0[3]);
            p.s[4] = f2bf(x1[0]); p.s[5] = f2bf(x1[1]);
            p.s[6] = f2bf(x1[2]); p.s[7] = f2bf(x1[3]);
            alds[row * ROWU4 + 64 + c] = p.u;
        }
        __builtin_amdgcn_s_wait_asynccnt(0);
        __syncthreads();

        // prefetch next step's x slab while we compute (global_prefetch_b8)
        if (t + 1 < Tdim)
            __builtin_prefetch(x + ((long)(tid >> 1) * Tdim + (t + 1)) * Idim + (tid & 1) * 128, 0, 0);

        // ---- accumulators: 4 gates x (16x16 f32 tile) initialized to bias ----
        v8f acc[4];
#pragma unroll
        for (int g = 0; g < 4; ++g) {
#pragma unroll
            for (int r = 0; r < 8; ++r) acc[g][r] = brs[g];
        }

        // ---- K loop: 24 x (1 A-frag load + 4 B-frag loads + 4 WMMA) ----
        const int arow = wave * 16 + l16;
#pragma unroll 4
        for (int kk = 0; kk < Kdim / 32; ++kk) {
            FragBF a;   // A 16x32 bf16: lane holds K-chunks [k0,k0+8) and [k0+16,k0+24)
            a.u[0] = alds[arow * ROWU4 + kk * 4 + lhalf];
            a.u[1] = alds[arow * ROWU4 + kk * 4 + lhalf + 2];
#pragma unroll
            for (int g = 0; g < 4; ++g) {
                FragBF b; // B 32x16 bf16: lane holds K-chunks [k0,k0+8) and [k0+16,k0+24) for col l16
                int gn = g * 16 + l16;
                b.u[0] = wlds[gn * ROWU4 + kk * 4 + lhalf];
                b.u[1] = wlds[gn * ROWU4 + kk * 4 + lhalf + 2];
                acc[g] = __builtin_amdgcn_wmma_f32_16x16x32_bf16(
                    false, a.v, false, b.v, (short)0, acc[g], false, false);
            }
        }

        // ---- LSTM cell epilogue; c in registers, h -> global bf16 ----
        const int colg = bid * 16 + l16;
#pragma unroll
        for (int r = 0; r < 8; ++r) {
            int row = wave * 16 + r + lhalf * 8;      // C/D layout per ISA 7.12.2
            float iv = sigm(acc[0][r]);
            float fv = sigm(acc[1][r]);
            float gv = tanh_(acc[2][r]);
            float ov = sigm(acc[3][r]);
            float cn = fv * creg[r] + iv * gv;
            creg[r] = cn;
            hstate[row * Hdim + colg] = f2bf(ov * tanh_(cn));
        }

        // ---- device-wide step barrier (h must be visible to all 32 blocks) ----
        __threadfence();
        __syncthreads();
        if (tid == 0) {
            unsigned gen = __hip_atomic_load(&bar[1], __ATOMIC_ACQUIRE, __HIP_MEMORY_SCOPE_AGENT);
            unsigned arr = __hip_atomic_fetch_add(&bar[0], 1u, __ATOMIC_ACQ_REL, __HIP_MEMORY_SCOPE_AGENT);
            if (arr == NBLK - 1) {
                __hip_atomic_store(&bar[0], 0u, __ATOMIC_RELAXED, __HIP_MEMORY_SCOPE_AGENT);
                __hip_atomic_fetch_add(&bar[1], 1u, __ATOMIC_RELEASE, __HIP_MEMORY_SCOPE_AGENT);
            } else {
                while (__hip_atomic_load(&bar[1], __ATOMIC_ACQUIRE, __HIP_MEMORY_SCOPE_AGENT) == gen)
                    __builtin_amdgcn_s_sleep(2);
            }
        }
        __syncthreads();
    }
}

// ---------------- final fc: out[b] = hT[b,:] . fc_w + fc_b ----------------
__global__ void k_fc(const unsigned short* __restrict__ h,
                     const float* __restrict__ fcw, const float* __restrict__ fcb,
                     float* __restrict__ out) {
    int b = blockIdx.x * blockDim.x + threadIdx.x;
    if (b < Bdim) {
        float s = 0.0f;
        for (int j = 0; j < Hdim; ++j) s += bf2f(h[b * Hdim + j]) * fcw[j];
        out[b] = s + fcb[0];
    }
}

// ---------------- launch ----------------
extern "C" void kernel_launch(void* const* d_in, const int* in_sizes, int n_in,
                              void* d_out, int out_size, void* d_ws, size_t ws_size,
                              hipStream_t stream) {
    const float* x   = (const float*)d_in[0];
    const float* Wih = (const float*)d_in[1];
    const float* Whh = (const float*)d_in[2];
    const float* bih = (const float*)d_in[3];
    const float* bhh = (const float*)d_in[4];
    const float* fcw = (const float*)d_in[5];
    const float* fcb = (const float*)d_in[6];
    float* out = (float*)d_out;
    char*  ws  = (char*)d_ws;

    unsigned*       bar    = (unsigned*)ws;
    unsigned short* hstate = (unsigned short*)(ws + OFF_H);
    float*          bias   = (float*)(ws + OFF_BIAS);
    unsigned short* wcat   = (unsigned short*)(ws + OFF_WCAT);

    k_init<<<32, 256, 0, stream>>>(bar, (uv4*)hstate);
    k_prep_w<<<1024, 256, 0, stream>>>(Wih, Whh, bih, bhh, wcat, bias);

    size_t smem = (size_t)(64 + 128) * ROWU4 * 16;   // 297,984 B < 320 KB WGP LDS
    (void)hipFuncSetAttribute((const void*)k_lstm,
                              hipFuncAttributeMaxDynamicSharedMemorySize, (int)smem);
    k_lstm<<<NBLK, THREADS, smem, stream>>>(x, wcat, bias, hstate, bar);

    k_fc<<<1, 128, 0, stream>>>(hstate, fcw, fcb, out);
}
